// S2ConvNet_original_19567871001312
// MI455X (gfx1250) — matrix operand order repaired and checked by reading, not compiled, not run on hardware
//
#include <hip/hip_runtime.h>
#include <math.h>

typedef __attribute__((ext_vector_type(16))) _Float16 v16h;
typedef __attribute__((ext_vector_type(8)))  float    v8f;

#define DEV __device__ __forceinline__

// ---------------- static problem sizes ----------------
constexpr int NB    = 128;           // batch
constexpr int F1c   = 20;            // features after conv1
constexpr int F2c   = 40;            // features after conv2
constexpr int FOUTc = 10;
constexpr int BIN   = 30;            // input bandwidth
constexpr int B1cc  = 10;            // bandwidth after conv1
constexpr int B2cc  = 6;             // bandwidth after conv2
constexpr int NA1   = 60;            // 2*BIN
constexpr int NA2   = 20;            // 2*B1
constexpr int NA3   = 12;            // 2*B2
constexpr int S1c   = 100;           // sum (2l+1), l<10
constexpr int S2c   = 286;           // sum (2l+1)^2, l<6
constexpr int NG1   = 24;            // s2 kernel grid points
constexpr int NG2   = 192;           // so3 kernel grid points

constexpr double PI_D = 3.14159265358979323846;

__constant__ int d_SQBASE[10] = {0,1,10,35,84,165,286,455,680,969}; // prefix of (2l+1)^2

// ---------------- workspace layout (float elements) ----------------
constexpr int OFF_A1   = 0;
constexpr int OFF_Y1R  = OFF_A1   + NA1*S1c;
constexpr int OFF_Y1I  = OFF_Y1R  + NG1*S1c;
constexpr int OFF_D1S  = OFF_Y1I  + NG1*S1c;
constexpr int OFF_W3   = OFF_D1S  + NA2*1330;
constexpr int OFF_Y2R  = OFF_W3   + NA2*S2c;
constexpr int OFF_Y2I  = OFF_Y2R  + NG2*S2c;
constexpr int OFF_D2S  = OFF_Y2I  + NG2*S2c;
constexpr int OFF_WINT = OFF_D2S  + NA3*S2c;
constexpr int OFF_M1I  = OFF_WINT + NA3;     // int table
constexpr int OFF_M2I  = OFF_M1I  + S1c;     // int table
constexpr int OFF_N2I  = OFF_M2I  + S2c;     // int table
constexpr int OFF_XFR  = OFF_N2I  + S2c;
constexpr int XFN      = NB*NA1*NA1;
constexpr int OFF_XFI  = OFF_XFR  + XFN;
constexpr int OFF_XHR  = OFF_XFI  + XFN;
constexpr int OFF_XHI  = OFF_XHR  + S1c*NB;
constexpr int OFF_K1HR = OFF_XHI  + S1c*NB;
constexpr int OFF_K1HI = OFF_K1HR + S1c*F1c;
constexpr int OFF_BIGR = OFF_K1HI + S1c*F1c;
constexpr int BIGN     = NB*F1c*NA2*NA2*NA2;       // 20,480,000
constexpr int OFF_BIGI = OFF_BIGR + BIGN;
constexpr int OFF_X1   = OFF_BIGI + BIGN;
constexpr int OFF_XH2R = OFF_X1   + BIGN;
constexpr int XH2N     = S2c*NB*F1c;
constexpr int OFF_XH2I = OFF_XH2R + XH2N;
constexpr int OFF_K2HR = OFF_XH2I + XH2N;
constexpr int K2HN     = S2c*F1c*F2c;
constexpr int OFF_K2HI = OFF_K2HR + K2HN;
constexpr int OFF_ZLR  = OFF_K2HI + K2HN;
constexpr int ZLN      = NB*F2c*286;               // 5120 * sum L^2
constexpr int OFF_ZLI  = OFF_ZLR  + ZLN;
constexpr int PXF      = (1408*224)/2;             // f16 panel (floats)
constexpr int OFF_PXR  = OFF_ZLI  + ZLN;
constexpr int OFF_PXI  = OFF_PXR  + PXF;
constexpr int PGF      = (448*224)/2;
constexpr int OFF_PGR  = OFF_PXI  + PXF;
constexpr int OFF_PGI  = OFF_PGR  + PGF;
constexpr int OFF_PGN  = OFF_PGI  + PGF;           // pre-negated Gi panel
constexpr int X2N      = NB*F2c*NA3*NA3*NA3;       // 8,847,360
constexpr int OFF_X2   = OFF_PGN  + PGF;
constexpr int WS_TOTAL = OFF_X2   + X2N;

// ---------------- math helpers (table precompute, f64) ----------------
DEV double dfact(int n){ double r=1.0; for(int i=2;i<=n;++i) r*=(double)i; return r; }

DEV double wigd(int l,int mp,int m,double beta){
  double cb=cos(0.5*beta), sb=sin(0.5*beta);
  double pref=sqrt(dfact(l+mp)*dfact(l-mp)*dfact(l+m)*dfact(l-m));
  int s0=(m-mp)>0?(m-mp):0;
  int s1=(l+m)<(l-mp)?(l+m):(l-mp);
  double acc=0.0;
  for(int s=s0;s<=s1;++s){
    double den=dfact(l+m-s)*dfact(s)*dfact(mp-m+s)*dfact(l-mp-s);
    double sg=((mp-m+s)&1)?-1.0:1.0;
    acc += sg/den * pow(cb,(double)(2*l+m-mp-2*s)) * pow(sb,(double)(mp-m+2*s));
  }
  return pref*acc;
}

DEV double dhw(int b,int j){
  double beta=PI_D*(2*j+1)/(4.0*b);
  double s=0.0;
  for(int k=0;k<b;++k) s += sin((2*k+1)*beta)/(double)(2*k+1);
  return (2.0/b)*sin(beta)*s;
}

DEV void decode_s1(int s,int&l,int&m){ l=0; while(s>=(l+1)*(l+1)) ++l; m=s-l*l-l; }
DEV void decode_s2(int p,int&l,int&mi,int&ni){
  int rem=p; l=0;
  for(;;){ int L=2*l+1; if(rem<L*L){ mi=rem/L; ni=rem%L; return; } rem-=L*L; ++l; }
}
DEV void s2grid(int g,double&beta,double&alpha){
  int ib=g/8, ia=g%8;
  beta=(ib+1)*(PI_D/8.0)/3.0; alpha=2.0*PI_D*ia/8.0;
}
DEV void so3grid(int g,double&beta,double&alpha,double&gamma){
  int ib=g/64, r=g%64, ia=r/8, ig=r%8;
  beta=(ib+1)*(PI_D/8.0)/3.0; alpha=2.0*PI_D*ia/8.0; gamma=2.0*PI_D*ig/8.0-alpha;
}

// ---------------- table kernels ----------------
__global__ void kt_A1(float* ws){
  int id=blockIdx.x*blockDim.x+threadIdx.x; if(id>=NA1*S1c) return;
  int j=id/S1c, s=id%S1c, l,m; decode_s1(s,l,m);
  double beta=PI_D*(2*j+1)/(4.0*BIN);
  ws[OFF_A1+id]=(float)(wigd(l,m,0,beta)*dhw(BIN,j)*(2.0*PI_D/(double)NA1));
}
__global__ void kt_Y1(float* ws){
  int id=blockIdx.x*blockDim.x+threadIdx.x; if(id>=NG1*S1c) return;
  int g=id/S1c, s=id%S1c, l,m; decode_s1(s,l,m);
  double beta,alpha; s2grid(g,beta,alpha);
  double d=wigd(l,m,0,beta), ph=(double)m*alpha;
  ws[OFF_Y1R+id]=(float)( d*cos(ph));
  ws[OFF_Y1I+id]=(float)(-d*sin(ph));
}
__global__ void kt_D1S(float* ws){
  int id=blockIdx.x*blockDim.x+threadIdx.x; if(id>=NA2*1330) return;
  int rem=id, l=0;
  for(;;){ int sz=NA2*(2*l+1)*(2*l+1); if(rem<sz) break; rem-=sz; ++l; }
  int L=2*l+1, j=rem/(L*L), r2=rem%(L*L), mi=r2/L, ni=r2%L;
  double beta=PI_D*(2*j+1)/(4.0*B1cc);
  ws[OFF_D1S+id]=(float)((double)L*wigd(l,mi-l,ni-l,beta));
}
__global__ void kt_W3(float* ws){
  int id=blockIdx.x*blockDim.x+threadIdx.x; if(id>=NA2*S2c) return;
  int j=id/S2c, p=id%S2c, l,mi,ni; decode_s2(p,l,mi,ni);
  double beta=PI_D*(2*j+1)/(4.0*B1cc);
  double w=dhw(B1cc,j)*(2.0*PI_D/(double)NA2)*(2.0*PI_D/(double)NA2);
  ws[OFF_W3+id]=(float)(wigd(l,mi-l,ni-l,beta)*w);
}
__global__ void kt_Y2(float* ws){
  int id=blockIdx.x*blockDim.x+threadIdx.x; if(id>=NG2*S2c) return;
  int g=id/S2c, p=id%S2c, l,mi,ni; decode_s2(p,l,mi,ni);
  double beta,alpha,gamma; so3grid(g,beta,alpha,gamma);
  double d=wigd(l,mi-l,ni-l,beta);
  double ph=(double)(mi-l)*alpha+(double)(ni-l)*gamma;
  ws[OFF_Y2R+id]=(float)( d*cos(ph));
  ws[OFF_Y2I+id]=(float)(-d*sin(ph));
}
__global__ void kt_D2S(float* ws){
  int id=blockIdx.x*blockDim.x+threadIdx.x; if(id>=NA3*S2c) return;
  int rem=id, l=0;
  for(;;){ int sz=NA3*(2*l+1)*(2*l+1); if(rem<sz) break; rem-=sz; ++l; }
  int L=2*l+1, j=rem/(L*L), r2=rem%(L*L), mi=r2/L, ni=r2%L;
  double beta=PI_D*(2*j+1)/(4.0*B2cc);
  ws[OFF_D2S+id]=(float)((double)L*wigd(l,mi-l,ni-l,beta));
}
__global__ void kt_WINT(float* ws){
  int j=blockIdx.x*blockDim.x+threadIdx.x; if(j>=NA3) return;
  ws[OFF_WINT+j]=(float)(dhw(B2cc,j)*(2.0*PI_D/(double)NA3)*(2.0*PI_D/(double)NA3));
}
__global__ void kt_IDX(float* ws){
  int id=blockIdx.x*blockDim.x+threadIdx.x;
  if(id<S1c){
    int l,m; decode_s1(id,l,m);
    ((int*)(ws+OFF_M1I))[id]=((m%NA1)+NA1)%NA1;
  } else if(id<S1c+S2c){
    int p=id-S1c, l,mi,ni; decode_s2(p,l,mi,ni);
    ((int*)(ws+OFF_M2I))[p]=(((mi-l)%NA2)+NA2)%NA2;
  } else if(id<S1c+2*S2c){
    int p=id-S1c-S2c, l,mi,ni; decode_s2(p,l,mi,ni);
    ((int*)(ws+OFF_N2I))[p]=(((ni-l)%NA2)+NA2)%NA2;
  }
}

// ---------------- stage kernels ----------------
// FFT over alpha (60-point direct DFT): xf[b,j,s]
__global__ void k_fft60(const float* __restrict__ x, float* ws){
  int id=blockIdx.x*blockDim.x+threadIdx.x; if(id>=XFN) return;
  int b=id/(NA1*NA1), r=id%(NA1*NA1), j=r/NA1, s=r%NA1;
  const float* row=x+((size_t)b*NA1+j)*NA1;
  float ar=0.f, ai=0.f;
  for(int a=0;a<NA1;++a){
    int k=(s*a)%NA1;
    float th=-(float)(2.0*PI_D/NA1)*(float)k, c,sn;
    __sincosf(th,&sn,&c);
    ar += row[a]*c; ai += row[a]*sn;
  }
  ws[OFF_XFR+id]=ar; ws[OFF_XFI+id]=ai;
}
// S2 spectrum: xhat[s,b] = sum_j xf[b,j,M1IDX[s]] * A1[j,s]
__global__ void k_xhat(float* ws){
  int id=blockIdx.x*blockDim.x+threadIdx.x; if(id>=S1c*NB) return;
  int s=id/NB, b=id%NB;
  int idx=((const int*)(ws+OFF_M1I))[s];
  float ar=0.f, ai=0.f;
  for(int j=0;j<NA1;++j){
    float a1=ws[OFF_A1+j*S1c+s];
    int src=(b*NA1+j)*NA1+idx;
    ar += a1*ws[OFF_XFR+src]; ai += a1*ws[OFF_XFI+src];
  }
  ws[OFF_XHR+id]=ar; ws[OFF_XHI+id]=ai;
}
// kernel1 spectrum: k1h[s,o] = SC1 * sum_p k1[o,p]*Y1[p,s]
__global__ void k_k1h(const float* __restrict__ k1, float* ws){
  int id=blockIdx.x*blockDim.x+threadIdx.x; if(id>=S1c*F1c) return;
  int s=id/F1c, o=id%F1c;
  float sc=(float)(1.0/sqrt((double)NG1*1.0*10000.0/900.0));
  float ar=0.f, ai=0.f;
  for(int p=0;p<NG1;++p){
    float kv=k1[o*NG1+p];
    ar += kv*ws[OFF_Y1R+p*S1c+s];
    ai += kv*ws[OFF_Y1I+p*S1c+s];
  }
  ws[OFF_K1HR+id]=ar*sc; ws[OFF_K1HI+id]=ai*sc;
}
// conv1: fused degree-sum + inverse gamma-DFT -> G1[b,o,j,mm,g]
__global__ void k_g1(float* ws){
  int id=blockIdx.x*blockDim.x+threadIdx.x; if(id>=BIGN) return;
  int g=id%NA2, mm=(id/NA2)%NA2, j=(id/(NA2*NA2))%NA2, o=(id/(NA2*NA2*NA2))%F1c, b=id/(F1c*NA2*NA2*NA2);
  int m=(mm<10)?mm:mm-20;
  int am=m<0?-m:m;
  float ar=0.f, ai=0.f;
  if(am<=9){
    for(int l=am;l<10;++l){
      int L=2*l+1;
      int slm=l*l+l+m;
      float xr=ws[OFF_XHR+slm*NB+b], xi=ws[OFF_XHI+slm*NB+b];
      int doff=OFF_D1S+NA2*d_SQBASE[l]+(j*L+(m+l))*L;
      for(int n=-l;n<=l;++n){
        int sln=l*l+l+n;
        float kr=ws[OFF_K1HR+sln*F1c+o], ki=ws[OFF_K1HI+sln*F1c+o];
        float zr=xr*kr+xi*ki, zi=xi*kr-xr*ki;        // xhat * conj(k1h)
        float d=ws[doff+(n+l)];
        int kk=((n*g)%NA2+NA2)%NA2;
        float th=(float)(2.0*PI_D/NA2)*(float)kk, c,sn;
        __sincosf(th,&sn,&c);
        ar += d*(zr*c-zi*sn);
        ai += d*(zr*sn+zi*c);
      }
    }
  }
  ws[OFF_BIGR+id]=ar; ws[OFF_BIGI+id]=ai;
}
// conv1: inverse alpha-DFT + bias + relu -> x1[b,o,j,a,g]
__global__ void k_x1(const float* __restrict__ bias1, float* ws){
  int id=blockIdx.x*blockDim.x+threadIdx.x; if(id>=BIGN) return;
  int g=id%NA2, a=(id/NA2)%NA2, j=(id/(NA2*NA2))%NA2, o=(id/(NA2*NA2*NA2))%F1c, b=id/(F1c*NA2*NA2*NA2);
  int base=(((b*F1c+o)*NA2+j)*NA2)*NA2;
  float acc=0.f;
  for(int mm=0;mm<NA2;++mm){
    int kk=(mm*a)%NA2;
    float th=(float)(2.0*PI_D/NA2)*(float)kk, c,sn;
    __sincosf(th,&sn,&c);
    acc += ws[OFF_BIGR+base+mm*NA2+g]*c - ws[OFF_BIGI+base+mm*NA2+g]*sn;
  }
  acc += bias1[o];
  ws[OFF_X1+id]=acc>0.f?acc:0.f;
}
// conv2 analysis pass1: H1[b,f,j,m,g] = sum_a x1 * e^{-2pi i m a/20}
__global__ void k_h1(float* ws){
  int id=blockIdx.x*blockDim.x+threadIdx.x; if(id>=BIGN) return;
  int g=id%NA2, m=(id/NA2)%NA2, j=(id/(NA2*NA2))%NA2, f=(id/(NA2*NA2*NA2))%F1c, b=id/(F1c*NA2*NA2*NA2);
  int base=(((b*F1c+f)*NA2+j)*NA2)*NA2;
  float hr=0.f, hi=0.f;
  for(int a=0;a<NA2;++a){
    int kk=(m*a)%NA2;
    float th=(float)(2.0*PI_D/NA2)*(float)kk, c,sn;
    __sincosf(th,&sn,&c);
    float xv=ws[OFF_X1+base+a*NA2+g];
    hr += xv*c; hi -= xv*sn;
  }
  ws[OFF_BIGR+id]=hr; ws[OFF_BIGI+id]=hi;
}
// conv2 analysis: xhat2[p,b,f] = sum_j W3[j,p] * sum_g H1[b,f,j,mI,g] e^{-2pi i nI g/20}
__global__ void k_xhat2(float* ws){
  int id=blockIdx.x*blockDim.x+threadIdx.x; if(id>=XH2N) return;
  int p=id/(NB*F1c), r=id%(NB*F1c), b=r/F1c, f=r%F1c;
  int mI=((const int*)(ws+OFF_M2I))[p];
  int nI=((const int*)(ws+OFF_N2I))[p];
  float ar=0.f, ai=0.f;
  for(int j=0;j<NA2;++j){
    float wv=ws[OFF_W3+j*S2c+p];
    int base=(((b*F1c+f)*NA2+j)*NA2+mI)*NA2;
    float sr=0.f, si=0.f;
    for(int g=0;g<NA2;++g){
      int kk=(nI*g)%NA2;
      float th=(float)(2.0*PI_D/NA2)*(float)kk, c,sn;
      __sincosf(th,&sn,&c);
      float hr=ws[OFF_BIGR+base+g], hi=ws[OFF_BIGI+base+g];
      sr += hr*c+hi*sn;      // (hr+ihi)*e^{-i th}
      si += hi*c-hr*sn;
    }
    ar += wv*sr; ai += wv*si;
  }
  ws[OFF_XH2R+id]=ar; ws[OFF_XH2I+id]=ai;
}
// kernel2 spectrum: k2h[p,i,o] = SC2 * sum_g k2[i,o,g]*Y2[g,p]
__global__ void k_k2h(const float* __restrict__ k2, float* ws){
  int id=blockIdx.x*blockDim.x+threadIdx.x; if(id>=K2HN) return;
  int p=id/(F1c*F2c), q=id%(F1c*F2c);
  float sc=(float)(1.0/sqrt((double)NG2*(double)F1c*216.0/1000.0));
  float ar=0.f, ai=0.f;
  for(int g=0;g<NG2;++g){
    float kv=k2[q*NG2+g];
    ar += kv*ws[OFF_Y2R+g*S2c+p];
    ai += kv*ws[OFF_Y2I+g*S2c+p];
  }
  ws[OFF_K2HR+p*(F1c*F2c)+q]=ar*sc;
  ws[OFF_K2HI+p*(F1c*F2c)+q]=ai*sc;
}
// pack X panel (f16, K-major) for degree l:  row=(b,m), K=(k,i)
__global__ void k_pack_x(float* ws, int l){
  int L=2*l+1, Mr=NB*L, K=F1c*L, Kp=(K+31)&~31;
  int id=blockIdx.x*blockDim.x+threadIdx.x; if(id>=Mr*Kp) return;
  int row=id/Kp, kk=id%Kp;
  float vr=0.f, vi=0.f;
  if(kk<K){
    int b=row/L, m=row%L, k=kk/F1c, i=kk%F1c;
    int src=(d_SQBASE[l]+m*L+k)*(NB*F1c)+b*F1c+i;
    vr=ws[OFF_XH2R+src]; vi=ws[OFF_XH2I+src];
  }
  ((_Float16*)(ws+OFF_PXR))[id]=(_Float16)vr;
  ((_Float16*)(ws+OFF_PXI))[id]=(_Float16)vi;
}
// pack G panels (conj(k2h), f16, K-major, pre-transposed): row=(n,o), K=(k,i)
// PGR = Re(k2h), PGI = -Im(k2h) (conj), PGN = +Im(k2h) (negated conj, for -Xi*Gi)
__global__ void k_pack_g(float* ws, int l){
  int L=2*l+1, Nn=F2c*L, K=F1c*L, Kp=(K+31)&~31, Np=(Nn+15)&~15;
  int id=blockIdx.x*blockDim.x+threadIdx.x; if(id>=Np*Kp) return;
  int row=id/Kp, kk=id%Kp;
  float vr=0.f, vi=0.f;
  if(row<Nn && kk<K){
    int n=row/F2c, o=row%F2c, k=kk/F1c, i=kk%F1c;
    int src=(d_SQBASE[l]+n*L+k)*(F1c*F2c)+i*F2c+o;
    vr= ws[OFF_K2HR+src];
    vi=-ws[OFF_K2HI+src];   // conjugate
  }
  ((_Float16*)(ws+OFF_PGR))[id]=(_Float16)vr;
  ((_Float16*)(ws+OFF_PGI))[id]=(_Float16)vi;
  ((_Float16*)(ws+OFF_PGN))[id]=(_Float16)(-vi);
}
// Fused complex WMMA GEMM:
//   Zr[M,N] = Xr*G^T(r) - Xi*G^T(i) ; Zi[M,N] = Xr*G^T(i) + Xi*G^T(r)
// 4 waves / block: each wave owns one 16-row M tile; shared 32x16 G tiles staged in LDS
// via 16-byte vector copies (global_load_b128 -> ds_store_b128).
__global__ void k_cgemm(const _Float16* __restrict__ Xr, const _Float16* __restrict__ Xi,
                        const _Float16* __restrict__ Gr, const _Float16* __restrict__ Gi,
                        const _Float16* __restrict__ Gn,
                        float* __restrict__ Zr, float* __restrict__ Zi,
                        int M, int N, int Kp, int ldc)
{
  __shared__ alignas(16) _Float16 sGr[16*32];
  __shared__ alignas(16) _Float16 sGi[16*32];
  __shared__ alignas(16) _Float16 sGn[16*32];
  int tid=threadIdx.x;
  int lane=tid&31, wave=tid>>5;
  int half=lane>>4, l16=lane&15;
  int tm=blockIdx.x*64+wave*16, tn=blockIdx.y*16;
  const _Float16* xrp=Xr+(size_t)(tm+l16)*Kp;
  const _Float16* xip=Xi+(size_t)(tm+l16)*Kp;
  // staging assignment: 96 32-byte chunks = 3 panels * 16 rows * 2 chunks
  const _Float16* gsrc0=0; _Float16* gdst0=0;
  int srow=0, schunk=0;
  if(tid<96){
    int pnl=tid/32, rc=tid%32;
    srow=rc>>1; schunk=rc&1;
    gsrc0=(pnl==0)?Gr:((pnl==1)?Gi:Gn);
    gdst0=(pnl==0)?sGr:((pnl==1)?sGi:sGn);
  }
  v8f accR={}, accI={};
  for(int k0=0;k0<Kp;k0+=32){
    __syncthreads();
    if(tid<96){
      const uint4* src=(const uint4*)(gsrc0+(size_t)(tn+srow)*Kp+k0+schunk*16);
      uint4* dst=(uint4*)(gdst0+srow*32+schunk*16);
      dst[0]=src[0]; dst[1]=src[1];
    }
    __syncthreads();
    const _Float16* a0=xrp+k0+half*8;
    const _Float16* a1=xip+k0+half*8;
    __builtin_prefetch(a0+64,0,0);
    __builtin_prefetch(a1+64,0,0);
    const _Float16* g0=&sGr[l16*32+half*8];
    const _Float16* g1=&sGi[l16*32+half*8];
    const _Float16* g2=&sGn[l16*32+half*8];
    v16h vxr,vxi,vgr,vgi,vgn;
    #pragma unroll
    for(int e=0;e<8;++e){
      vxr[e]=a0[e]; vxr[e+8]=a0[e+16];
      vxi[e]=a1[e]; vxi[e+8]=a1[e+16];
      vgr[e]=g0[e]; vgr[e+8]=g0[e+16];
      vgi[e]=g1[e]; vgi[e+8]=g1[e+16];
      vgn[e]=g2[e]; vgn[e+8]=g2[e+16];
    }
    accR=__builtin_amdgcn_wmma_f32_16x16x32_f16(false,vxr,false,vgr,(short)0,accR,false,false);
    accR=__builtin_amdgcn_wmma_f32_16x16x32_f16(false,vxi,false,vgn,(short)0,accR,false,false);
    accI=__builtin_amdgcn_wmma_f32_16x16x32_f16(false,vxr,false,vgi,(short)0,accI,false,false);
    accI=__builtin_amdgcn_wmma_f32_16x16x32_f16(false,vxi,false,vgr,(short)0,accI,false,false);
  }
  #pragma unroll
  for(int r=0;r<8;++r){
    int row=tm+r+8*half, col=tn+l16;
    if(row<M && col<N){
      Zr[(size_t)row*ldc+col]=accR[r];
      Zi[(size_t)row*ldc+col]=accI[r];
    }
  }
}
// conv2 synthesis: fused degree-sum + inverse gamma-DFT -> G2[b,o,j,mm,g]
__global__ void k_g2(float* ws){
  int id=blockIdx.x*blockDim.x+threadIdx.x; if(id>=X2N) return;
  int g=id%NA3, mm=(id/NA3)%NA3, j=(id/(NA3*NA3))%NA3, o=(id/(NA3*NA3*NA3))%F2c, b=id/(F2c*NA3*NA3*NA3);
  int m=(mm<6)?mm:mm-12;
  int am=m<0?-m:m;
  float ar=0.f, ai=0.f;
  if(am<=5){
    for(int l=am;l<6;++l){
      int L=2*l+1;
      int zoff=5120*d_SQBASE[l];
      int doff=OFF_D2S+NA3*d_SQBASE[l]+(j*L+(m+l))*L;
      int zrow=(b*L+(m+l))*(F2c*L);
      for(int n=-l;n<=l;++n){
        int col=(n+l)*F2c+o;
        float zr=ws[OFF_ZLR+zoff+zrow+col], zi=ws[OFF_ZLI+zoff+zrow+col];
        float d=ws[doff+(n+l)];
        int kk=((n*g)%NA3+NA3)%NA3;
        float th=(float)(2.0*PI_D/NA3)*(float)kk, c,sn;
        __sincosf(th,&sn,&c);
        ar += d*(zr*c-zi*sn);
        ai += d*(zr*sn+zi*c);
      }
    }
  }
  ws[OFF_BIGR+id]=ar; ws[OFF_BIGI+id]=ai;   // reuse BIG region
}
// conv2: inverse alpha-DFT + bias + relu -> x2[b,o,j,a,g]
__global__ void k_x2(const float* __restrict__ bias2, float* ws){
  int id=blockIdx.x*blockDim.x+threadIdx.x; if(id>=X2N) return;
  int g=id%NA3, a=(id/NA3)%NA3, j=(id/(NA3*NA3))%NA3, o=(id/(NA3*NA3*NA3))%F2c, b=id/(F2c*NA3*NA3*NA3);
  int base=(((b*F2c+o)*NA3+j)*NA3)*NA3;
  float acc=0.f;
  for(int mm=0;mm<NA3;++mm){
    int kk=(mm*a)%NA3;
    float th=(float)(2.0*PI_D/NA3)*(float)kk, c,sn;
    __sincosf(th,&sn,&c);
    acc += ws[OFF_BIGR+base+mm*NA3+g]*c - ws[OFF_BIGI+base+mm*NA3+g]*sn;
  }
  acc += bias2[o];
  ws[OFF_X2+id]=acc>0.f?acc:0.f;
}
// so3 integrate + linear head
__global__ void k_head(const float* __restrict__ wout, const float* __restrict__ blin,
                       float* __restrict__ out, float* ws){
  int id=blockIdx.x*blockDim.x+threadIdx.x; if(id>=NB*FOUTc) return;
  int b=id/FOUTc, c=id%FOUTc;
  float acc=blin[c];
  for(int f=0;f<F2c;++f){
    float feat=0.f;
    int base=(b*F2c+f)*(NA3*NA3*NA3);
    for(int j=0;j<NA3;++j){
      float wj=ws[OFF_WINT+j];
      float s=0.f;
      int jb=base+j*(NA3*NA3);
      for(int q=0;q<NA3*NA3;++q) s += ws[OFF_X2+jb+q];
      feat += wj*s;
    }
    acc += wout[c*F2c+f]*feat;
  }
  out[id]=acc;
}

// ---------------- launcher ----------------
extern "C" void kernel_launch(void* const* d_in, const int* in_sizes, int n_in,
                              void* d_out, int out_size, void* d_ws, size_t ws_size,
                              hipStream_t stream)
{
  const float* x    =(const float*)d_in[0];
  const float* k1   =(const float*)d_in[1];
  const float* bias1=(const float*)d_in[2];
  const float* k2   =(const float*)d_in[3];
  const float* bias2=(const float*)d_in[4];
  const float* wout =(const float*)d_in[5];
  const float* blin =(const float*)d_in[6];
  float* out=(float*)d_out;
  float* ws =(float*)d_ws;
  if(ws_size < (size_t)WS_TOTAL*sizeof(float)) return;

  auto gs=[](int n){ return dim3((unsigned)((n+255)/256)); };

  // tables
  kt_A1  <<<gs(NA1*S1c),256,0,stream>>>(ws);
  kt_Y1  <<<gs(NG1*S1c),256,0,stream>>>(ws);
  kt_D1S <<<gs(NA2*1330),256,0,stream>>>(ws);
  kt_W3  <<<gs(NA2*S2c),256,0,stream>>>(ws);
  kt_Y2  <<<gs(NG2*S2c),256,0,stream>>>(ws);
  kt_D2S <<<gs(NA3*S2c),256,0,stream>>>(ws);
  kt_WINT<<<gs(NA3),256,0,stream>>>(ws);
  kt_IDX <<<gs(S1c+2*S2c),256,0,stream>>>(ws);

  // conv1
  k_fft60<<<gs(XFN),256,0,stream>>>(x,ws);
  k_xhat <<<gs(S1c*NB),256,0,stream>>>(ws);
  k_k1h  <<<gs(S1c*F1c),256,0,stream>>>(k1,ws);
  k_g1   <<<gs(BIGN),256,0,stream>>>(ws);
  k_x1   <<<gs(BIGN),256,0,stream>>>(bias1,ws);

  // conv2 analysis
  k_h1   <<<gs(BIGN),256,0,stream>>>(ws);
  k_xhat2<<<gs(XH2N),256,0,stream>>>(ws);
  k_k2h  <<<gs(K2HN),256,0,stream>>>(k2,ws);

  // conv2 per-degree fused complex GEMMs via WMMA (LDS-shared B operand)
  static const int SQB[10]={0,1,10,35,84,165,286,455,680,969};
  const _Float16* Xr=(const _Float16*)(ws+OFF_PXR);
  const _Float16* Xi=(const _Float16*)(ws+OFF_PXI);
  const _Float16* Gr=(const _Float16*)(ws+OFF_PGR);
  const _Float16* Gi=(const _Float16*)(ws+OFF_PGI);
  const _Float16* Gn=(const _Float16*)(ws+OFF_PGN);
  for(int l=0;l<6;++l){
    int L=2*l+1, Mr=NB*L, Nn=F2c*L, K=F1c*L;
    int Kp=(K+31)&~31, Np=(Nn+15)&~15;
    k_pack_x<<<gs(Mr*Kp),256,0,stream>>>(ws,l);
    k_pack_g<<<gs(Np*Kp),256,0,stream>>>(ws,l);
    float* Zr=ws+OFF_ZLR+5120*SQB[l];
    float* Zi=ws+OFF_ZLI+5120*SQB[l];
    dim3 gg((unsigned)(Mr/64),(unsigned)(Np/16));   // Mr=128*(2l+1) is a multiple of 64
    k_cgemm<<<gg,128,0,stream>>>(Xr,Xi,Gr,Gi,Gn,Zr,Zi,Mr,Nn,Kp,Nn);
  }

  // conv2 synthesis + head
  k_g2  <<<gs(X2N),256,0,stream>>>(ws);
  k_x2  <<<gs(X2N),256,0,stream>>>(bias2,ws);
  k_head<<<gs(NB*FOUTc),256,0,stream>>>(wout,blin,out,ws);
}
